// Seq2seq_86062554677655
// MI455X (gfx1250) — matrix operand order repaired and checked by reading, not compiled, not run on hardware
//
#include <hip/hip_runtime.h>
#include <hip/hip_bf16.h>

#define B_    64
#define T_    400
#define E_    300
#define EPAD  320
#define HE_   512
#define HD_   1024
#define V_    50000
#define DEC_  80
#define NT_   3125      /* V/16 column tiles of logits */
#define NCHUNK_ 391     /* ceil(3125/8) — 8 tiles per workgroup */

typedef __attribute__((ext_vector_type(16))) __bf16 bf16x16;
typedef __attribute__((ext_vector_type(8)))  float  f32x8;

// ---------------------------------------------------------------- helpers ---
__device__ __forceinline__ __bf16 f2bf(float f) {
  unsigned u = __float_as_uint(f);
  unsigned r = (u + 0x7FFFu + ((u >> 16) & 1u)) >> 16;
  unsigned short s = (unsigned short)r;
  __bf16 b;
  __builtin_memcpy(&b, &s, 2);
  return b;
}

__device__ __forceinline__ float sigf(float x) { return 1.0f / (1.0f + __expf(-x)); }

// A-matrix 16x32 bf16 fragment: lane l holds row (l&15); K = k0+(l>=16?8:0)+{0..7}
// in elems 0..7 and K+16 in elems 8..15 (ISA 7.12.2).
__device__ __forceinline__ bf16x16 loadA(const __bf16* __restrict__ rowptr, int k0, int lane) {
  int off = k0 + ((lane & 16) ? 8 : 0);
  union { bf16x16 v; uint4 q[2]; } u;
  u.q[0] = *(const uint4*)(rowptr + off);
  u.q[1] = *(const uint4*)(rowptr + off + 16);
  return u.v;
}

// B-matrix 32x16 bf16 fragment: lane l holds column (l&15); K = k0+(l>=16?8:0)+{0..7}
// in elems 0..7 and K+16 in elems 8..15 (same as A).
__device__ __forceinline__ bf16x16 loadB(const __bf16* __restrict__ colptr, int k0, int lane) {
  int off = k0 + ((lane & 16) ? 8 : 0);
  union { bf16x16 v; uint4 q[2]; } u;
  u.q[0] = *(const uint4*)(colptr + off);
  u.q[1] = *(const uint4*)(colptr + off + 16);
  return u.v;
}

__device__ __forceinline__ f32x8 wmma_bf(f32x8 c, bf16x16 a, bf16x16 b) {
  return __builtin_amdgcn_wmma_f32_16x16x32_bf16(false, a, false, b, (short)0, c, false, false);
}

__device__ __forceinline__ unsigned ordf(float f) {
  unsigned u = __float_as_uint(f);
  return (u & 0x80000000u) ? ~u : (u | 0x80000000u);
}

__device__ __forceinline__ unsigned long long shflxor64(unsigned long long v, int m) {
  unsigned lo = (unsigned)v, hi = (unsigned)(v >> 32);
  lo = __shfl_xor(lo, m, 32);
  hi = __shfl_xor(hi, m, 32);
  return (((unsigned long long)hi) << 32) | lo;
}

// ------------------------------------------------------------ prep kernels ---
__global__ void k_cvt_pad(const float* __restrict__ src, __bf16* __restrict__ dst,
                          int R, int C, int CP) {
  long long i = (long long)blockIdx.x * 256 + threadIdx.x;
  long long n = (long long)R * CP;
  if (i >= n) return;
  int r = (int)(i / CP), c = (int)(i % CP);
  float v = (c < C) ? src[(long long)r * C + c] : 0.0f;
  dst[i] = f2bf(v);
}

__global__ void k_zero_h(float* hF, __bf16* hFb, float* hB, __bf16* hBb) {
  int i = blockIdx.x * 256 + threadIdx.x;
  if (i < B_ * HE_) {
    hF[i] = 0.0f; hB[i] = 0.0f;
    hFb[i] = f2bf(0.0f); hBb[i] = f2bf(0.0f);
  }
}

__global__ void k_dec_init(const float* __restrict__ hF, const float* __restrict__ hB,
                           float* hD, __bf16* hDb, int* tok) {
  int i = blockIdx.x * 256 + threadIdx.x;
  if (i >= B_ * HD_) return;
  int b = i >> 10, c = i & 1023;
  float v = (c < HE_) ? hF[b * HE_ + c] : hB[b * HE_ + (c - HE_)];
  hD[i] = v; hDb[i] = f2bf(v);
  if (i < B_) tok[i] = 1;  // SOS
}

// ------------------------------------------------------ encoder GRU step ----
// grid: (16,1,2) x 256.  blockIdx.z: 0=fwd, 1=bwd.  128 waves per dir:
// wave = mt*32+nt, mt in 0..3 (batch tiles), nt in 0..31 (H tiles).
__global__ __launch_bounds__(256) void k_enc_step(
    const int* __restrict__ texts, const int* __restrict__ lens,
    const __bf16* __restrict__ emb,
    const __bf16* __restrict__ wihF, const __bf16* __restrict__ whhF,
    const float* __restrict__ bihF, const float* __restrict__ bhhF,
    const __bf16* __restrict__ wihB, const __bf16* __restrict__ whhB,
    const float* __restrict__ bihB, const float* __restrict__ bhhB,
    const float* hFc, const __bf16* hFcb, float* hFn, __bf16* hFnb,
    const float* hBc, const __bf16* hBcb, float* hBn, __bf16* hBnb,
    int tf, int tb) {
  const int dir = blockIdx.z;
  const int t = dir ? tb : tf;
  const __bf16* wih = dir ? wihB : wihF;
  const __bf16* whh = dir ? whhB : whhF;
  const float* bih = dir ? bihB : bihF;
  const float* bhh = dir ? bhhB : bhhF;
  const float* hc  = dir ? hBc : hFc;
  const __bf16* hcb = dir ? hBcb : hFcb;
  float*  hout  = dir ? hBn : hFn;
  __bf16* houtb = dir ? hBnb : hFnb;

  const int lane = threadIdx.x & 31;
  const int wave = blockIdx.x * 8 + (threadIdx.x >> 5);  // 0..127
  const int mt = wave >> 5, nt = wave & 31;
  const int m0 = mt << 4, n0 = nt << 4;
  const int arow = m0 + (lane & 15);
  const int bn = n0 + (lane & 15);

  f32x8 cir = {0,0,0,0,0,0,0,0}, ciz = cir, cin = cir;
  f32x8 chr = cir, chz = cir, chn = cir;

  // gi = x @ wih.T  (x = emb[texts[:,t]], K = EPAD)
  const __bf16* xrow = emb + (size_t)texts[arow * T_ + t] * EPAD;
  const __bf16* br = wih + (size_t)bn * EPAD;
  const __bf16* bz = wih + (size_t)(HE_ + bn) * EPAD;
  const __bf16* bnn = wih + (size_t)(2 * HE_ + bn) * EPAD;
  for (int k = 0; k < EPAD; k += 32) {
    bf16x16 a = loadA(xrow, k, lane);
    cir = wmma_bf(cir, a, loadB(br, k, lane));
    ciz = wmma_bf(ciz, a, loadB(bz, k, lane));
    cin = wmma_bf(cin, a, loadB(bnn, k, lane));
  }
  // gh = h @ whh.T  (K = HE)
  const __bf16* hrow = hcb + (size_t)arow * HE_;
  const __bf16* wr = whh + (size_t)bn * HE_;
  const __bf16* wz = whh + (size_t)(HE_ + bn) * HE_;
  const __bf16* wn = whh + (size_t)(2 * HE_ + bn) * HE_;
  for (int k = 0; k < HE_; k += 32) {
    bf16x16 a = loadA(hrow, k, lane);
    chr = wmma_bf(chr, a, loadB(wr, k, lane));
    chz = wmma_bf(chz, a, loadB(wz, k, lane));
    chn = wmma_bf(chn, a, loadB(wn, k, lane));
  }

  const int col = n0 + (lane & 15);
  const float bir = bih[col], biz = bih[HE_ + col], bin_ = bih[2 * HE_ + col];
  const float bhr = bhh[col], bhz = bhh[HE_ + col], bhn = bhh[2 * HE_ + col];
  const int rbase = m0 + ((lane & 16) ? 8 : 0);
#pragma unroll
  for (int r = 0; r < 8; r++) {
    int row = rbase + r;
    float rg = sigf(cir[r] + bir + chr[r] + bhr);
    float zg = sigf(ciz[r] + biz + chz[r] + bhz);
    float ng = tanhf(cin[r] + bin_ + rg * (chn[r] + bhn));
    float ho = hc[(size_t)row * HE_ + col];
    float hv = (1.0f - zg) * ng + zg * ho;
    float o = (t < lens[row]) ? hv : ho;
    hout[(size_t)row * HE_ + col] = o;
    houtb[(size_t)row * HE_ + col] = f2bf(o);
  }
}

// ------------------------------------------------------ decoder GRU step ----
// grid: 32 x 256. 256 waves: mt in 0..3, nt in 0..63.
__global__ __launch_bounds__(256) void k_dec_step(
    const int* __restrict__ tok, const __bf16* __restrict__ emb,
    const __bf16* __restrict__ wih, const __bf16* __restrict__ whh,
    const float* __restrict__ bih, const float* __restrict__ bhh,
    const float* __restrict__ hc, const __bf16* __restrict__ hcb,
    float* hout, __bf16* houtb) {
  const int lane = threadIdx.x & 31;
  const int wave = blockIdx.x * 8 + (threadIdx.x >> 5);  // 0..255
  const int mt = wave >> 6, nt = wave & 63;
  const int m0 = mt << 4, n0 = nt << 4;
  const int arow = m0 + (lane & 15);
  const int bn = n0 + (lane & 15);

  f32x8 cir = {0,0,0,0,0,0,0,0}, ciz = cir, cin = cir;
  f32x8 chr = cir, chz = cir, chn = cir;

  const __bf16* xrow = emb + (size_t)tok[arow] * EPAD;
  const __bf16* br = wih + (size_t)bn * EPAD;
  const __bf16* bz = wih + (size_t)(HD_ + bn) * EPAD;
  const __bf16* bnn = wih + (size_t)(2 * HD_ + bn) * EPAD;
  for (int k = 0; k < EPAD; k += 32) {
    bf16x16 a = loadA(xrow, k, lane);
    cir = wmma_bf(cir, a, loadB(br, k, lane));
    ciz = wmma_bf(ciz, a, loadB(bz, k, lane));
    cin = wmma_bf(cin, a, loadB(bnn, k, lane));
  }
  const __bf16* hrow = hcb + (size_t)arow * HD_;
  const __bf16* wr = whh + (size_t)bn * HD_;
  const __bf16* wz = whh + (size_t)(HD_ + bn) * HD_;
  const __bf16* wn = whh + (size_t)(2 * HD_ + bn) * HD_;
  for (int k = 0; k < HD_; k += 32) {
    bf16x16 a = loadA(hrow, k, lane);
    chr = wmma_bf(chr, a, loadB(wr, k, lane));
    chz = wmma_bf(chz, a, loadB(wz, k, lane));
    chn = wmma_bf(chn, a, loadB(wn, k, lane));
  }

  const int col = n0 + (lane & 15);
  const float bir = bih[col], biz = bih[HD_ + col], bin_ = bih[2 * HD_ + col];
  const float bhr = bhh[col], bhz = bhh[HD_ + col], bhn = bhh[2 * HD_ + col];
  const int rbase = m0 + ((lane & 16) ? 8 : 0);
#pragma unroll
  for (int r = 0; r < 8; r++) {
    int row = rbase + r;
    float rg = sigf(cir[r] + bir + chr[r] + bhr);
    float zg = sigf(ciz[r] + biz + chz[r] + bhz);
    float ng = tanhf(cin[r] + bin_ + rg * (chn[r] + bhn));
    float ho = hc[(size_t)row * HD_ + col];
    float hv = (1.0f - zg) * ng + zg * ho;
    hout[(size_t)row * HD_ + col] = hv;
    houtb[(size_t)row * HD_ + col] = f2bf(hv);
  }
}

// -------------------------------------- logits GEMM + fused partial argmax --
// grid: (NCHUNK_, 4) x 256. Each wave: one 16x16 tile of h @ out_w.T (+bias),
// then per-row max packed as (ordf(val)<<32)|~col, reduced lane-wise + LDS.
__global__ __launch_bounds__(256) void k_logits(
    const __bf16* __restrict__ hDb, const __bf16* __restrict__ outw,
    const float* __restrict__ outb, unsigned long long* __restrict__ partial) {
  __shared__ unsigned long long part[16][8];
  const int lane = threadIdx.x & 31;
  const int w = threadIdx.x >> 5;
  const int chunk = blockIdx.x;
  const int mt = blockIdx.y;
  const int ntile = chunk * 8 + w;

  unsigned long long best[8];
#pragma unroll
  for (int r = 0; r < 8; r++) best[r] = 0ull;

  if (ntile < NT_) {
    const __bf16* hrow = hDb + (size_t)(mt * 16 + (lane & 15)) * HD_;
    const int colb = ntile * 16 + (lane & 15);
    const __bf16* wrow = outw + (size_t)colb * HD_;
    f32x8 c = {0,0,0,0,0,0,0,0};
    for (int k = 0; k < HD_; k += 32)
      c = wmma_bf(c, loadA(hrow, k, lane), loadB(wrow, k, lane));
    const float bb = outb[colb];
#pragma unroll
    for (int r = 0; r < 8; r++) {
      float v = c[r] + bb;
      best[r] = (((unsigned long long)ordf(v)) << 32) | (unsigned)(~colb);
    }
    // reduce over the 16 columns held by each half-group of lanes
    for (int m = 1; m < 16; m <<= 1) {
#pragma unroll
      for (int r = 0; r < 8; r++) {
        unsigned long long o = shflxor64(best[r], m);
        if (o > best[r]) best[r] = o;
      }
    }
  }
  if ((lane & 15) == 0) {
    int half = lane >> 4;
#pragma unroll
    for (int r = 0; r < 8; r++) part[half * 8 + r][w] = best[r];
  }
  __syncthreads();
  if (threadIdx.x < 16) {
    unsigned long long b0 = part[threadIdx.x][0];
    for (int w2 = 1; w2 < 8; w2++) {
      unsigned long long o = part[threadIdx.x][w2];
      if (o > b0) b0 = o;
    }
    partial[(size_t)(mt * 16 + threadIdx.x) * NCHUNK_ + chunk] = b0;
  }
}

// final per-row argmax over NCHUNK_ partials; emits token + output value
__global__ void k_argmax_fin(const unsigned long long* __restrict__ partial,
                             int* tok, float* out, int step) {
  __shared__ unsigned long long s[128];
  const int row = blockIdx.x;
  unsigned long long b = 0ull;
  for (int c = threadIdx.x; c < NCHUNK_; c += 128) {
    unsigned long long o = partial[(size_t)row * NCHUNK_ + c];
    if (o > b) b = o;
  }
  s[threadIdx.x] = b;
  __syncthreads();
  for (int st = 64; st > 0; st >>= 1) {
    if (threadIdx.x < st) {
      unsigned long long o = s[threadIdx.x + st];
      if (o > s[threadIdx.x]) s[threadIdx.x] = o;
    }
    __syncthreads();
  }
  if (threadIdx.x == 0) {
    unsigned idx = ~(unsigned)(s[0] & 0xFFFFFFFFull);
    tok[row] = (int)idx;
    out[row * DEC_ + step] = (float)idx;
  }
}

// ------------------------------------------------------------------ driver --
extern "C" void kernel_launch(void* const* d_in, const int* in_sizes, int n_in,
                              void* d_out, int out_size, void* d_ws, size_t ws_size,
                              hipStream_t stream) {
  const int*   texts = (const int*)d_in[0];
  const int*   lens  = (const int*)d_in[1];
  const float* emb   = (const float*)d_in[2];
  const float* wihF  = (const float*)d_in[3];
  const float* whhF  = (const float*)d_in[4];
  const float* bihF  = (const float*)d_in[5];
  const float* bhhF  = (const float*)d_in[6];
  const float* wihB  = (const float*)d_in[7];
  const float* whhB  = (const float*)d_in[8];
  const float* bihB  = (const float*)d_in[9];
  const float* bhhB  = (const float*)d_in[10];
  const float* dwih  = (const float*)d_in[11];
  const float* dwhh  = (const float*)d_in[12];
  const float* dbih  = (const float*)d_in[13];
  const float* dbhh  = (const float*)d_in[14];
  const float* outw  = (const float*)d_in[15];
  const float* outb  = (const float*)d_in[16];
  float* out = (float*)d_out;

  char* p = (char*)d_ws;
  auto alloc = [&](size_t bytes) -> char* {
    char* r = p;
    p += (bytes + 255) & ~(size_t)255;
    return r;
  };
  __bf16* emb_b  = (__bf16*)alloc((size_t)V_ * EPAD * 2);
  __bf16* wihF_b = (__bf16*)alloc((size_t)3 * HE_ * EPAD * 2);
  __bf16* wihB_b = (__bf16*)alloc((size_t)3 * HE_ * EPAD * 2);
  __bf16* whhF_b = (__bf16*)alloc((size_t)3 * HE_ * HE_ * 2);
  __bf16* whhB_b = (__bf16*)alloc((size_t)3 * HE_ * HE_ * 2);
  __bf16* dwih_b = (__bf16*)alloc((size_t)3 * HD_ * EPAD * 2);
  __bf16* dwhh_b = (__bf16*)alloc((size_t)3 * HD_ * HD_ * 2);
  __bf16* outw_b = (__bf16*)alloc((size_t)V_ * HD_ * 2);
  float* hF[2]; __bf16* hFb[2]; float* hB[2]; __bf16* hBb[2];
  for (int i = 0; i < 2; i++) { hF[i] = (float*)alloc(B_ * HE_ * 4); hFb[i] = (__bf16*)alloc(B_ * HE_ * 2); }
  for (int i = 0; i < 2; i++) { hB[i] = (float*)alloc(B_ * HE_ * 4); hBb[i] = (__bf16*)alloc(B_ * HE_ * 2); }
  float* hD[2]; __bf16* hDb[2];
  for (int i = 0; i < 2; i++) { hD[i] = (float*)alloc(B_ * HD_ * 4); hDb[i] = (__bf16*)alloc(B_ * HD_ * 2); }
  int* tok = (int*)alloc(B_ * 4);
  unsigned long long* partial = (unsigned long long*)alloc((size_t)B_ * NCHUNK_ * 8);

  auto cvt = [&](const float* s, __bf16* d, int R, int C, int CP) {
    long long n = (long long)R * CP;
    int g = (int)((n + 255) / 256);
    k_cvt_pad<<<dim3(g), dim3(256), 0, stream>>>(s, d, R, C, CP);
  };
  cvt(emb,  emb_b,  V_,      E_,  EPAD);
  cvt(wihF, wihF_b, 3 * HE_, E_,  EPAD);
  cvt(wihB, wihB_b, 3 * HE_, E_,  EPAD);
  cvt(whhF, whhF_b, 3 * HE_, HE_, HE_);
  cvt(whhB, whhB_b, 3 * HE_, HE_, HE_);
  cvt(dwih, dwih_b, 3 * HD_, E_,  EPAD);
  cvt(dwhh, dwhh_b, 3 * HD_, HD_, HD_);
  cvt(outw, outw_b, V_,      HD_, HD_);

  k_zero_h<<<dim3((B_ * HE_ + 255) / 256), dim3(256), 0, stream>>>(hF[0], hFb[0], hB[0], hBb[0]);

  int c = 0;
  for (int t = 0; t < T_; t++) {
    k_enc_step<<<dim3(16, 1, 2), dim3(256), 0, stream>>>(
        texts, lens, emb_b,
        wihF_b, whhF_b, bihF, bhhF,
        wihB_b, whhB_b, bihB, bhhB,
        hF[c], hFb[c], hF[1 - c], hFb[1 - c],
        hB[c], hBb[c], hB[1 - c], hBb[1 - c],
        t, T_ - 1 - t);
    c ^= 1;
  }

  k_dec_init<<<dim3((B_ * HD_ + 255) / 256), dim3(256), 0, stream>>>(hF[c], hB[c], hD[0], hDb[0], tok);

  int d = 0;
  for (int s = 0; s < DEC_; s++) {
    k_dec_step<<<dim3(32), dim3(256), 0, stream>>>(
        tok, emb_b, dwih_b, dwhh_b, dbih, dbhh,
        hD[d], hDb[d], hD[1 - d], hDb[1 - d]);
    k_logits<<<dim3(NCHUNK_, 4), dim3(256), 0, stream>>>(hDb[1 - d], outw_b, outb, partial);
    k_argmax_fin<<<dim3(B_), dim3(128), 0, stream>>>(partial, tok, out, s);
    d ^= 1;
  }
}
